// Decoder_85469849190866
// MI455X (gfx1250) — matrix operand order repaired and checked by reading, not compiled, not run on hardware
//
#include <hip/hip_runtime.h>
#include <hip/hip_bf16.h>
#include <math.h>

// ---------------------------------------------------------------------------
// LSTM decoder for MI455X (gfx1250): bf16 WMMA (f32 accum) step kernels.
//   B=128, H=512, 4H=2048, T=512, OUT=33
// Step kernel grid: 16 WGs x 256 threads (8 waves). WG owns 32 hidden columns
// across all 4 gates -> fused LSTM cell. Wave owns a 128x16 gate strip.
// CDNA5 paths: v_wmma_f32_16x16x32_bf16, TDM tensor_load_to_lds (TENSORcnt,
// hardware LDS pad for the 520-stride layout), s_wait_tensorcnt,
// global_prefetch_b8.
// ---------------------------------------------------------------------------

#define BATCH 128
#define HID   512
#define GATES 2048          // 4*HID
#define TLEN  512
#define NOUT  33
#define BH    (BATCH*HID)   // 65536
#define GH    (GATES*HID)   // 1048576 elems per weight matrix per layer
#define XS_STRIDE 520       // LDS row stride in bf16 (pad vs bank conflicts)
#define STEP_LDS_BYTES (BATCH*XS_STRIDE*2)   // 133120 B (epilogue reuses it)

typedef __bf16 bf16;
typedef __attribute__((ext_vector_type(16))) __bf16 v16bf;
typedef __attribute__((ext_vector_type(8)))  __bf16 v8bf;
typedef __attribute__((ext_vector_type(8)))  float  v8f;
typedef __attribute__((ext_vector_type(4)))  unsigned u32x4;
typedef __attribute__((ext_vector_type(8)))  unsigned u32x8;

__device__ __forceinline__ float sigmoidf_(float x) {
    return 1.0f / (1.0f + __expf(-x));
}

// B tile: lane = (n, khalf); 16 contiguous K values from one W row.
__device__ __forceinline__ v16bf load_b16(const bf16* p) {
    union { v16bf v; v8bf h[2]; } u;
    u.h[0] = *(const v8bf*)(p);
    u.h[1] = *(const v8bf*)(p + 8);
    return u.v;
}
// A tile: lane = (m, khalf); same K order as B: khalf=0 lanes hold K 0-7 & 8-15,
// khalf=1 lanes hold K 16-23 & 24-31  -> two 8-element chunks.
__device__ __forceinline__ v16bf load_a16(const bf16* row, int kA, int kB) {
    union { v16bf v; v8bf h[2]; } u;
    u.h[0] = *(const v8bf*)(row + kA);
    u.h[1] = *(const v8bf*)(row + kB);
    return u.v;
}

// TDM: DMA a 2D tile (rows x cols, 2-byte elems, global row stride = srcStride
// elems) into LDS at lds_addr, inserting padDw DWORDs of LDS padding after
// every 1024 bytes (pad_interval code 7) -> hardware-applied 520-elem stride.
// D# per CDNA5 ISA; 2D => VADDR2/3 NULL.
__device__ __forceinline__ void tdm_load_tile_2d(unsigned lds_addr,
                                                 const void* gsrc,
                                                 unsigned cols, unsigned rows,
                                                 unsigned srcStride,
                                                 unsigned padDwCode) {
    const unsigned long long ga = (unsigned long long)(uintptr_t)gsrc;
    u32x4 g0;
    g0.x = 1u;                                   // count=1, user descriptor
    g0.y = lds_addr;                             // lds_addr[31:0]
    g0.z = (unsigned)ga;                         // global_addr[31:0]
    g0.w = ((unsigned)(ga >> 32) & 0x01FFFFFFu)  // global_addr[56:32]
           | 0x80000000u;                        // type=2 ("image")
    u32x8 g1;
    g1.s0 = (1u << 16)                           // data_size=1 (2 bytes)
          | (1u << 20)                           // pad_enable
          | (7u << 22)                           // pad_interval: 256 DWORDs
          | (padDwCode << 25);                   // pad_amount: code 3 = 4 DW
    g1.s1 = (cols & 0xFFFFu) << 16;              // tensor_dim0[15:0] @ 63:48
    g1.s2 = (rows & 0xFFFFu) << 16;              // tensor_dim1[15:0] @ 95:80
    g1.s3 = (cols & 0xFFFFu) << 16;              // tile_dim0 @ 127:112
    g1.s4 = rows;                                // tile_dim1 @ 143:128
    g1.s5 = srcStride;                           // tensor_dim0_stride[31:0]
    g1.s6 = 0u;
    g1.s7 = 0u;
    asm volatile("tensor_load_to_lds %0, %1"
                 :: "s"(g0), "s"(g1)
                 : "memory");
}

// ---------------------------------------------------------------------------
// Prep: f32 -> bf16 weights, fused bias, state init (deterministic per call).
// ---------------------------------------------------------------------------
__global__ __launch_bounds__(256)
void prep_kernel(const float* __restrict__ Wih, const float* __restrict__ Whh,
                 const float* __restrict__ bih, const float* __restrict__ bhh,
                 const float* __restrict__ h0,  const float* __restrict__ c0,
                 bf16* __restrict__ Wih_bf, bf16* __restrict__ Whh_bf,
                 float* __restrict__ bias, bf16* __restrict__ hbuf,
                 float* __restrict__ c_ws) {
    const int stride = gridDim.x * blockDim.x;
    const int g0 = blockIdx.x * blockDim.x + threadIdx.x;
    for (int i = g0; i < 2 * GH; i += stride) {
        Wih_bf[i] = (bf16)Wih[i];
        Whh_bf[i] = (bf16)Whh[i];
    }
    for (int i = g0; i < 2 * GATES; i += stride)
        bias[i] = bih[i] + bhh[i];
    for (int i = g0; i < 2 * BH; i += stride) {
        const int layer = i >> 16;           // i / BH
        const int r     = i & (BH - 1);
        hbuf[layer * 2 * BH + r] = (bf16)h0[i];   // parity 0
        c_ws[i] = c0[i];
    }
}

// ---------------------------------------------------------------------------
// One LSTM timestep for one layer. LAYER==0 builds x_t from emb/acts/durs/cond;
// LAYER==1 stages h0_next via one TDM descriptor (hardware LDS padding).
// Gates GEMM via v_wmma_f32_16x16x32_bf16, fused cell epilogue,
// ping-pong h, in-place c.
// ---------------------------------------------------------------------------
template<int LAYER>
__global__ __launch_bounds__(256)
void lstm_step(int t,
               const bf16*  __restrict__ Wih,     // [GATES*HID] this layer
               const bf16*  __restrict__ Whh,     // [GATES*HID] this layer
               const float* __restrict__ bias,    // [GATES]     this layer
               const bf16*  __restrict__ a_src,   // LAYER==1: h0_next
               const int*   __restrict__ acts,    // [B,T]
               const float* __restrict__ durs,    // [B,T]
               const float* __restrict__ emb,     // [32,511]
               const float* __restrict__ cond,    // [B,HID]
               const bf16*  __restrict__ h_prev,  // [B,HID] bf16 (parity p)
               bf16*        __restrict__ h_next,  // [B,HID] bf16 (parity 1-p)
               float*       __restrict__ c_ws,    // [B,HID] f32 in-place
               bf16*        __restrict__ h1_seq)  // LAYER==1: + t*BH
{
    extern __shared__ char smem[];
    bf16*  xs   = (bf16*)smem;                    // [128][XS_STRIDE]
    float* gbuf = (float*)smem;                   // reused: [4][128][32] f32

    const int tid = threadIdx.x;
    const int wg  = blockIdx.x;                   // 0..15 (32 cols per gate)

    // ---- stage A-source (128 x 512 bf16, stride 520) into LDS ----
    if constexpr (LAYER == 0) {
        // x_t = [emb[act]+cond | dur+cond] gathered/built per element.
        for (int idx = tid; idx < BATCH * HID; idx += 256) {
            const int b = idx >> 9, k = idx & 511;
            float v;
            if (k < 511) {
                const int a = (t == 0) ? 0 : acts[b * TLEN + (t - 1)];
                v = emb[a * 511 + k] + cond[b * HID + k];
            } else {
                const float d = (t == 0) ? 0.0f : durs[b * TLEN + (t - 1)];
                v = d + cond[b * HID + k];
            }
            xs[b * XS_STRIDE + k] = (bf16)v;
        }
    } else {
        // One TDM DMA per workgroup (wave 0 issues; TDM ignores EXEC, may not
        // be clause'd). Hardware pads each 1024B row with 16B -> 520 stride.
        if (tid < 32) {
            tdm_load_tile_2d((unsigned)(uintptr_t)xs, a_src,
                             /*cols=*/HID, /*rows=*/BATCH,
                             /*srcStride=*/HID, /*padDwCode=*/3u);
            __builtin_amdgcn_s_wait_tensorcnt(0);
        }
    }
    __syncthreads();

    const int wave  = tid >> 5;
    const int lane  = tid & 31;
    const int gate  = wave >> 1;                     // 0..3 (i,f,g,o)
    const int ncol0 = wg * 32 + (wave & 1) * 16;     // column slice in gate
    const int nrow  = gate * HID + ncol0;            // row base in W
    const int nl    = lane & 15;
    const int kh    = lane >> 4;

    v8f acc[8] = {};

    const bf16* wb_ih = Wih + (size_t)(nrow + nl) * HID;
    const bf16* wb_hh = Whh + (size_t)(nrow + nl) * HID;

    // Warm the h-part weight stream while the x-part WMMAs run.
    __builtin_prefetch(wb_hh, 0, 0);
    __builtin_prefetch(wb_hh + 256, 0, 0);

    // ---- x-part: gates += X_t @ W_ih^T ----
    for (int k0 = 0; k0 < HID; k0 += 32) {
        const v16bf bt = load_b16(wb_ih + k0 + kh * 16);
        const int kA = k0 + kh * 16, kB = k0 + kh * 16 + 8;
#pragma unroll
        for (int mt = 0; mt < 8; ++mt) {
            const v16bf at = load_a16(xs + (mt * 16 + nl) * XS_STRIDE, kA, kB);
            acc[mt] = __builtin_amdgcn_wmma_f32_16x16x32_bf16(
                false, at, false, bt, (short)0, acc[mt], false, false);
        }
    }
    // ---- h-part: gates += h_prev @ W_hh^T (A direct from L2-resident global) ----
    for (int k0 = 0; k0 < HID; k0 += 32) {
        const v16bf bt = load_b16(wb_hh + k0 + kh * 16);
        const int kA = k0 + kh * 16, kB = k0 + kh * 16 + 8;
#pragma unroll
        for (int mt = 0; mt < 8; ++mt) {
            const v16bf at = load_a16(h_prev + (mt * 16 + nl) * HID, kA, kB);
            acc[mt] = __builtin_amdgcn_wmma_f32_16x16x32_bf16(
                false, at, false, bt, (short)0, acc[mt], false, false);
        }
    }

    __syncthreads();    // xs dead; reuse LDS as gbuf

    // ---- spill gate tiles (+bias) to LDS; C layout: lanes16-31 hold M=8..15 ----
    const float bv = bias[nrow + nl];
#pragma unroll
    for (int mt = 0; mt < 8; ++mt) {
#pragma unroll
        for (int r = 0; r < 8; ++r) {
            const int m = mt * 16 + r + (kh ? 8 : 0);
            gbuf[(gate * BATCH + m) * 32 + (wave & 1) * 16 + nl] = acc[mt][r] + bv;
        }
    }
    __syncthreads();

    // ---- fused LSTM cell over this WG's 128 x 32 slice ----
    for (int idx = tid; idx < BATCH * 32; idx += 256) {
        const int b = idx >> 5, j = idx & 31;
        const float iv = gbuf[(0 * BATCH + b) * 32 + j];
        const float fv = gbuf[(1 * BATCH + b) * 32 + j];
        const float gv = gbuf[(2 * BATCH + b) * 32 + j];
        const float ov = gbuf[(3 * BATCH + b) * 32 + j];
        const int hidx = b * HID + wg * 32 + j;
        const float c_new = sigmoidf_(fv) * c_ws[hidx] + sigmoidf_(iv) * tanhf(gv);
        const float h_new = sigmoidf_(ov) * tanhf(c_new);
        c_ws[hidx] = c_new;
        const bf16 hb = (bf16)h_new;
        h_next[hidx] = hb;
        if constexpr (LAYER == 1) h1_seq[hidx] = hb;
    }
}

// ---------------------------------------------------------------------------
// Output head: logits = h1 @ W_out^T + b, then log_softmax(32) | log_sigmoid.
// One block per (t,b); bandwidth-trivial (64 MB read / 8.6 MB write).
// ---------------------------------------------------------------------------
__global__ __launch_bounds__(64)
void out_proj(const bf16*  __restrict__ h1_seq,   // [T][B][HID]
              const float* __restrict__ Wout,     // [33][512]
              const float* __restrict__ bout,     // [33]
              float*       __restrict__ out)      // [B][T][33]
{
    __shared__ float hv[HID];
    __shared__ float lg[NOUT];
    const int bt = blockIdx.x;          // t*B + b (matches h1_seq storage)
    const int t = bt >> 7, b = bt & 127;
    const bf16* hp = h1_seq + (size_t)bt * HID;
    for (int k = threadIdx.x; k < HID; k += 64) hv[k] = (float)hp[k];
    __syncthreads();
    const int n = threadIdx.x;
    if (n < NOUT) {
        float s = bout[n];
        const float* wr = Wout + n * HID;
#pragma unroll 8
        for (int k = 0; k < HID; ++k) s = fmaf(hv[k], wr[k], s);
        lg[n] = s;
    }
    __syncthreads();
    float* op = out + ((size_t)b * TLEN + t) * NOUT;
    if (n < 32) {                        // wave 0: log_softmax over 32 classes
        const float x = lg[n];
        float m = x;
        for (int o = 16; o > 0; o >>= 1) m = fmaxf(m, __shfl_xor(m, o, 32));
        float se = __expf(x - m);
        for (int o = 16; o > 0; o >>= 1) se += __shfl_xor(se, o, 32);
        op[n] = x - m - __logf(se);
    } else if (n == 32) {                // lane 0 of wave 1: log_sigmoid
        const float x = lg[32];
        op[32] = fminf(x, 0.0f) - log1pf(__expf(-fabsf(x)));
    }
}

// ---------------------------------------------------------------------------
extern "C" void kernel_launch(void* const* d_in, const int* in_sizes, int n_in,
                              void* d_out, int out_size, void* d_ws, size_t ws_size,
                              hipStream_t stream) {
    const float* h0   = (const float*)d_in[0];
    const float* c0   = (const float*)d_in[1];
    const float* cond = (const float*)d_in[2];
    const int*   acts = (const int*)  d_in[3];
    const float* durs = (const float*)d_in[4];
    const float* emb  = (const float*)d_in[5];
    const float* Wih  = (const float*)d_in[6];
    const float* Whh  = (const float*)d_in[7];
    const float* bih  = (const float*)d_in[8];
    const float* bhh  = (const float*)d_in[9];
    const float* Wout = (const float*)d_in[10];
    const float* bout = (const float*)d_in[11];

    // Workspace layout (~73 MB total)
    char* p = (char*)d_ws;
    bf16*  Wih_bf = (bf16*)p;  p += (size_t)2 * GH * sizeof(bf16);       // 4 MB
    bf16*  Whh_bf = (bf16*)p;  p += (size_t)2 * GH * sizeof(bf16);       // 4 MB
    float* bias   = (float*)p; p += (size_t)2 * GATES * sizeof(float);   // 16 KB
    float* c_ws   = (float*)p; p += (size_t)2 * BH * sizeof(float);      // 512 KB
    bf16*  hbuf   = (bf16*)p;  p += (size_t)4 * BH * sizeof(bf16);       // 512 KB [layer][parity]
    bf16*  h1_seq = (bf16*)p;  p += (size_t)TLEN * BH * sizeof(bf16);    // 64 MB

    prep_kernel<<<4096, 256, 0, stream>>>(Wih, Whh, bih, bhh, h0, c0,
                                          Wih_bf, Whh_bf, bias, hbuf, c_ws);

    for (int t = 0; t < TLEN; ++t) {
        const int pr = t & 1;
        bf16* h0_prev = hbuf + (size_t)(0 * 2 + pr)       * BH;
        bf16* h0_next = hbuf + (size_t)(0 * 2 + (1 - pr)) * BH;
        bf16* h1_prev = hbuf + (size_t)(1 * 2 + pr)       * BH;
        bf16* h1_next = hbuf + (size_t)(1 * 2 + (1 - pr)) * BH;

        lstm_step<0><<<16, 256, STEP_LDS_BYTES, stream>>>(
            t, Wih_bf, Whh_bf, bias, (const bf16*)nullptr,
            acts, durs, emb, cond, h0_prev, h0_next, c_ws, (bf16*)nullptr);

        lstm_step<1><<<16, 256, STEP_LDS_BYTES, stream>>>(
            t, Wih_bf + GH, Whh_bf + GH, bias + GATES, h0_next,
            acts, durs, emb, cond, h1_prev, h1_next, c_ws + BH,
            h1_seq + (size_t)t * BH);
    }

    out_proj<<<BATCH * TLEN, 64, 0, stream>>>(h1_seq, Wout, bout, (float*)d_out);
}
